// _MSDeformAttnWrapper_25735444038170
// MI455X (gfx1250) — compile-verified, matrix operand order checked by reading, not run on hardware
//
#include <hip/hip_runtime.h>
#include <math.h>

// ---------------- problem constants (from reference) ----------------
#define LQ_N 13294
#define LV_N 13294
#define B_N  2
#define C_N  256
#define NH_N 8
#define L_N  4
#define P_N  4
#define D_N  32   // C / NH

typedef __attribute__((ext_vector_type(16))) __bf16    bf16x16;
typedef __attribute__((ext_vector_type(8)))  float     f32x8;
typedef __attribute__((ext_vector_type(8)))  unsigned  u32x8;

// K index within a 16x32 bf16 A (or 32x16 B) fragment for dword-slot vg and
// lane-half (lanes 0-15 = half 0, lanes 16-31 = half 1), per CDNA5 ISA 7.12.2.
__device__ __forceinline__ int kpair(int vg, int half) {
    return ((vg < 4) ? (2 * vg) : (16 + 2 * (vg - 4))) + 8 * half;
}

__device__ __forceinline__ unsigned pack_bf16(__bf16 a, __bf16 b) {
    unsigned short ua = __builtin_bit_cast(unsigned short, a);
    unsigned short ub = __builtin_bit_cast(unsigned short, b);
    return (unsigned)ua | ((unsigned)ub << 16);
}

__device__ __forceinline__ float bf16_lo(unsigned u) {
    return (float)__builtin_bit_cast(__bf16, (unsigned short)(u & 0xffffu));
}
__device__ __forceinline__ float bf16_hi(unsigned u) {
    return (float)__builtin_bit_cast(__bf16, (unsigned short)(u >> 16));
}

// Split-precision bf16 MMA: acc += A*B with A = ah+al, B = bh+bl (drop al*bl).
__device__ __forceinline__ f32x8 wmma_bf16x2(f32x8 acc, bf16x16 ah, bf16x16 al,
                                             bf16x16 bh, bf16x16 bl) {
    acc = __builtin_amdgcn_wmma_f32_16x16x32_bf16(false, ah, false, bh, (short)0, acc, false, false);
    acc = __builtin_amdgcn_wmma_f32_16x16x32_bf16(false, ah, false, bl, (short)0, acc, false, false);
    acc = __builtin_amdgcn_wmma_f32_16x16x32_bf16(false, al, false, bh, (short)0, acc, false, false);
    return acc;
}

__device__ __forceinline__ bf16x16 ld_frag(const unsigned* p) {
    return __builtin_bit_cast(bf16x16, *(const u32x8*)p);
}

// Fill one thread's share of the packed A panel in LDS.
// Thread (ks = tid>>5, ln = tid&31) packs lds[ks][ln][0..7] from its A row.
__device__ __forceinline__ void fill_a_panel(const float* __restrict__ arow,
                                             unsigned (*lds_hi)[32][8],
                                             unsigned (*lds_lo)[32][8],
                                             int tid) {
    int ks = tid >> 5;
    int ln = tid & 31;
    int half = ln >> 4;
#pragma unroll
    for (int vg = 0; vg < 8; ++vg) {
        int k = ks * 32 + kpair(vg, half);
        float x0 = arow[k], x1 = arow[k + 1];
        __bf16 h0 = (__bf16)x0, h1 = (__bf16)x1;
        lds_hi[ks][ln][vg] = pack_bf16(h0, h1);
        lds_lo[ks][ln][vg] = pack_bf16((__bf16)(x0 - (float)h0),
                                       (__bf16)(x1 - (float)h1));
    }
}

// ======= prep kernel: pack a [256 x N] f32 weight into fragment-ordered =====
// ======= hi/lo bf16 dwords: layout [K/32][N/16][32 lanes][8 dwords]     =====
__global__ __launch_bounds__(256) void k_pack_w(const float* __restrict__ W, int N,
                                                unsigned* __restrict__ hi,
                                                unsigned* __restrict__ lo) {
    int id = blockIdx.x * 256 + threadIdx.x;
    int total = (C_N / 32) * (N / 16) * 32 * 8;
    if (id >= total) return;
    int vg = id & 7;
    int t  = id >> 3;
    int ln = t & 31;
    t >>= 5;
    int nt = t % (N / 16);
    int ks = t / (N / 16);
    int k   = ks * 32 + kpair(vg, ln >> 4);
    int col = nt * 16 + (ln & 15);
    float x0 = W[(size_t)k * N + col];
    float x1 = W[(size_t)(k + 1) * N + col];
    __bf16 h0 = (__bf16)x0, h1 = (__bf16)x1;
    hi[id] = pack_bf16(h0, h1);
    lo[id] = pack_bf16((__bf16)(x0 - (float)h0), (__bf16)(x1 - (float)h1));
}

// ====== kernel 1: val = value @ Wv + bv -> bf16 [B, NH, LV, D] ======
// one block per 16-row M-tile; A panel staged+packed in LDS; wave w owns
// n-tiles {2w, 2w+1}.
__global__ __launch_bounds__(256) void k_val_proj(const float* __restrict__ value,
                                                  const unsigned* __restrict__ wv_hi,
                                                  const unsigned* __restrict__ wv_lo,
                                                  const float* __restrict__ bv,
                                                  __bf16* __restrict__ val_ws) {
    __shared__ unsigned lds_hi[8][32][8] __attribute__((aligned(32)));
    __shared__ unsigned lds_lo[8][32][8] __attribute__((aligned(32)));
    const int M = B_N * LV_N;
    int m0 = blockIdx.x * 16;
    int tid = threadIdx.x;
    {
        int mA  = min(m0 + (tid & 15), M - 1);
        int pos = mA % LV_N, b = mA / LV_N;        // value is [LV, B, C]
        fill_a_panel(value + ((size_t)pos * B_N + b) * C_N, lds_hi, lds_lo, tid);
    }
    __syncthreads();

    int wave = tid >> 5, lane = tid & 31;
    int half = lane >> 4, rrow = lane & 15;
    int nt0 = wave * 2, nt1 = nt0 + 1;
    f32x8 acc0 = {}, acc1 = {};
#pragma unroll
    for (int ks = 0; ks < 8; ++ks) {
        bf16x16 ah = ld_frag(&lds_hi[ks][lane][0]);
        bf16x16 al = ld_frag(&lds_lo[ks][lane][0]);
        size_t o0 = ((size_t)(ks * 16 + nt0) * 32 + lane) * 8;
        acc0 = wmma_bf16x2(acc0, ah, al, ld_frag(wv_hi + o0), ld_frag(wv_lo + o0));
        size_t o1 = ((size_t)(ks * 16 + nt1) * 32 + lane) * 8;
        acc1 = wmma_bf16x2(acc1, ah, al, ld_frag(wv_hi + o1), ld_frag(wv_lo + o1));
    }
    auto store = [&](const f32x8& acc, int nt) {
        int n = nt * 16 + rrow;
        float bvn = bv[n];
        int hh = n >> 5, dd = n & 31;
#pragma unroll
        for (int r = 0; r < 8; ++r) {
            int m = m0 + r + 8 * half;
            if (m < M) {
                int b2 = m / LV_N, pos2 = m % LV_N;
                val_ws[(((size_t)b2 * NH_N + hh) * LV_N + pos2) * D_N + dd] =
                    (__bf16)(acc[r] + bvn);
            }
        }
    };
    store(acc0, nt0);
    store(acc1, nt1);
}

// == kernel 2 (fused): off = q@Woff+boff -> loc ; logits = q@Wattn+battn ==
// 24 n-tiles total (0..15 offsets, 16..23 attn); wave w owns {3w,3w+1,3w+2}.
__global__ __launch_bounds__(256) void k_qproj(const float* __restrict__ query,
                                               const unsigned* __restrict__ woff_hi,
                                               const unsigned* __restrict__ woff_lo,
                                               const unsigned* __restrict__ wattn_hi,
                                               const unsigned* __restrict__ wattn_lo,
                                               const float* __restrict__ boff,
                                               const float* __restrict__ battn,
                                               const float* __restrict__ rp,
                                               float* __restrict__ loc_ws,
                                               float* __restrict__ logit_ws) {
    __shared__ unsigned lds_hi[8][32][8] __attribute__((aligned(32)));
    __shared__ unsigned lds_lo[8][32][8] __attribute__((aligned(32)));
    const int M = B_N * LQ_N;
    int m0 = blockIdx.x * 16;
    int tid = threadIdx.x;
    {
        int mA = min(m0 + (tid & 15), M - 1);
        int qi = mA % LQ_N, b = mA / LQ_N;          // query is [LQ, B, C]
        fill_a_panel(query + ((size_t)qi * B_N + b) * C_N, lds_hi, lds_lo, tid);
    }
    __syncthreads();

    int wave = tid >> 5, lane = tid & 31;
    int half = lane >> 4, rrow = lane & 15;
    f32x8 acc[3] = {{}, {}, {}};
#pragma unroll
    for (int ks = 0; ks < 8; ++ks) {
        bf16x16 ah = ld_frag(&lds_hi[ks][lane][0]);
        bf16x16 al = ld_frag(&lds_lo[ks][lane][0]);
#pragma unroll
        for (int j = 0; j < 3; ++j) {
            int nti = wave * 3 + j;                  // wave-uniform
            const unsigned *bh, *bl;
            if (nti < 16) {                          // offset GEMM, N=256
                size_t o = ((size_t)(ks * 16 + nti) * 32 + lane) * 8;
                bh = woff_hi + o;  bl = woff_lo + o;
            } else {                                 // attn GEMM, N=128
                size_t o = ((size_t)(ks * 8 + (nti - 16)) * 32 + lane) * 8;
                bh = wattn_hi + o; bl = wattn_lo + o;
            }
            acc[j] = wmma_bf16x2(acc[j], ah, al, ld_frag(bh), ld_frag(bl));
        }
    }
    const int SH[4][2] = {{100, 100}, {50, 50}, {25, 25}, {13, 13}};
#pragma unroll
    for (int j = 0; j < 3; ++j) {
        int nti = wave * 3 + j;
        if (nti < 16) {
            int n = nti * 16 + rrow;
            int comp = n & 1, p = (n >> 1) & 3, l = (n >> 3) & 3, hh = n >> 5;
            float norm = (float)(comp ? SH[l][0] : SH[l][1]);   // x->W, y->H
            float bn = boff[n];
#pragma unroll
            for (int r = 0; r < 8; ++r) {
                int m = m0 + r + 8 * half;
                if (m < M) {
                    int b2 = m / LQ_N, q2 = m % LQ_N;
                    float off = acc[j][r] + bn;
                    float rpv = rp[(((size_t)b2 * LQ_N + q2) * L_N + l) * 2 + comp];
                    rpv = fminf(fmaxf(rpv, 1e-5f), 1.0f - 1e-5f);
                    loc_ws[((((size_t)b2 * LQ_N + q2) * NH_N + hh) * 16 + l * 4 + p) * 2 + comp] =
                        rpv + off / norm;
                }
            }
        } else {
            int n = (nti - 16) * 16 + rrow;
            int hh = n >> 4, lp = n & 15;
            float bn = battn[n];
#pragma unroll
            for (int r = 0; r < 8; ++r) {
                int m = m0 + r + 8 * half;
                if (m < M) {
                    int b2 = m / LQ_N, q2 = m % LQ_N;
                    logit_ws[(((size_t)b2 * LQ_N + q2) * NH_N + hh) * 16 + lp] = acc[j][r] + bn;
                }
            }
        }
    }
}

// == kernel 3: softmax + bilinear sampling + einsum -> tmp [B*LQ, C] ==
// one wave per (b, q, h); bf16 values: 2 channels per lane, lanes 0-15 do
// points 0-7 and lanes 16-31 do points 8-15; halves merged with shfl_xor.
__global__ __launch_bounds__(256) void k_sample(const __bf16* __restrict__ val_ws,
                                                const float* __restrict__ loc_ws,
                                                const float* __restrict__ logit_ws,
                                                float* __restrict__ tmp_ws) {
    const int TOT = B_N * LQ_N * NH_N;
    int gw = blockIdx.x * 8 + (threadIdx.x >> 5);
    if (gw >= TOT) return;
    int h  = gw % NH_N;
    int bq = gw / NH_N;           // b*LQ + q
    int b  = bq / LQ_N;
    int lane  = threadIdx.x & 31;
    int lhalf = lane >> 4;        // which 8 points this half-wave handles
    int lx    = lane & 15;        // channel-pair index (channels 2lx, 2lx+1)

    const float* lg = logit_ws + (size_t)gw * 16;
    float w[16], mx = -3.4e38f;
#pragma unroll
    for (int p = 0; p < 16; ++p) { w[p] = lg[p]; mx = fmaxf(mx, w[p]); }
    float s = 0.f;
#pragma unroll
    for (int p = 0; p < 16; ++p) { w[p] = __expf(w[p] - mx); s += w[p]; }
    float inv = 1.0f / s;

    const float* loc = loc_ws + (size_t)gw * 32;
    const __bf16* vb = val_ws + ((size_t)b * NH_N + h) * (size_t)LV_N * D_N;
    const int SH[4][2] = {{100, 100}, {50, 50}, {25, 25}, {13, 13}};
    const int ST[4]    = {0, 10000, 12500, 13125};

    float ax = 0.f, ay = 0.f;
#pragma unroll
    for (int pp = 0; pp < 8; ++pp) {
        int p = lhalf * 8 + pp;
        int l = p >> 2;
        int Hl = SH[l][0], Wl = SH[l][1];
        float x = loc[2 * p] * (float)Wl - 0.5f;
        float y = loc[2 * p + 1] * (float)Hl - 0.5f;
        float x0f = floorf(x), y0f = floorf(y);
        float wx = x - x0f, wy = y - y0f;
        int x0 = (int)x0f, y0 = (int)y0f;
        const __bf16* vl = vb + (size_t)ST[l] * D_N;
        float sx = 0.f, sy = 0.f;
#pragma unroll
        for (int dy = 0; dy < 2; ++dy) {
            int yi = y0 + dy;
            float wyc = dy ? wy : 1.0f - wy;
#pragma unroll
            for (int dx = 0; dx < 2; ++dx) {
                int xi = x0 + dx;
                float wxc = dx ? wx : 1.0f - wx;
                bool valid = (xi >= 0) & (xi < Wl) & (yi >= 0) & (yi < Hl);
                int yc = min(max(yi, 0), Hl - 1);
                int xc = min(max(xi, 0), Wl - 1);
                const unsigned* rowp =
                    (const unsigned*)(vl + (size_t)(yc * Wl + xc) * D_N);
                unsigned u = rowp[lx];
                float wt = valid ? (wyc * wxc) : 0.0f;
                sx += wt * bf16_lo(u);
                sy += wt * bf16_hi(u);
            }
        }
        float wp = w[p] * inv;
        ax += wp * sx;
        ay += wp * sy;
    }
    ax += __shfl_xor(ax, 16, 32);
    ay += __shfl_xor(ay, 16, 32);
    if (lhalf == 0) {
        float* o = tmp_ws + (size_t)bq * C_N + h * D_N + 2 * lx;
        o[0] = ax;
        o[1] = ay;
    }
}

// == kernel 4: out = tmp @ Wo + bo + q  (residual fused) -> d_out [LQ,B,C] ==
__global__ __launch_bounds__(256) void k_out_proj(const float* __restrict__ tmp_ws,
                                                  const unsigned* __restrict__ wo_hi,
                                                  const unsigned* __restrict__ wo_lo,
                                                  const float* __restrict__ bo,
                                                  const float* __restrict__ query,
                                                  float* __restrict__ out) {
    __shared__ unsigned lds_hi[8][32][8] __attribute__((aligned(32)));
    __shared__ unsigned lds_lo[8][32][8] __attribute__((aligned(32)));
    const int M = B_N * LQ_N;
    int m0 = blockIdx.x * 16;
    int tid = threadIdx.x;
    {
        int mA = min(m0 + (tid & 15), M - 1);
        fill_a_panel(tmp_ws + (size_t)mA * C_N, lds_hi, lds_lo, tid);
    }
    __syncthreads();

    int wave = tid >> 5, lane = tid & 31;
    int half = lane >> 4, rrow = lane & 15;
    int nt0 = wave * 2, nt1 = nt0 + 1;
    f32x8 acc0 = {}, acc1 = {};
#pragma unroll
    for (int ks = 0; ks < 8; ++ks) {
        bf16x16 ah = ld_frag(&lds_hi[ks][lane][0]);
        bf16x16 al = ld_frag(&lds_lo[ks][lane][0]);
        size_t o0 = ((size_t)(ks * 16 + nt0) * 32 + lane) * 8;
        acc0 = wmma_bf16x2(acc0, ah, al, ld_frag(wo_hi + o0), ld_frag(wo_lo + o0));
        size_t o1 = ((size_t)(ks * 16 + nt1) * 32 + lane) * 8;
        acc1 = wmma_bf16x2(acc1, ah, al, ld_frag(wo_hi + o1), ld_frag(wo_lo + o1));
    }
    auto store = [&](const f32x8& acc, int nt) {
        int n = nt * 16 + rrow;
        float bon = bo[n];
#pragma unroll
        for (int r = 0; r < 8; ++r) {
            int m = m0 + r + 8 * half;
            if (m < M) {
                int b2 = m / LQ_N, q2 = m % LQ_N;
                size_t oi = ((size_t)q2 * B_N + b2) * C_N + n;   // [LQ, B, C]
                out[oi] = acc[r] + bon + query[oi];
            }
        }
    };
    store(acc0, nt0);
    store(acc1, nt1);
}

// ---------------------------------------------------------------------------
extern "C" void kernel_launch(void* const* d_in, const int* in_sizes, int n_in,
                              void* d_out, int out_size, void* d_ws, size_t ws_size,
                              hipStream_t stream) {
    (void)in_sizes; (void)n_in; (void)out_size; (void)ws_size;
    const float* query = (const float*)d_in[0];   // [LQ, B, C]
    const float* rp    = (const float*)d_in[1];   // [B, LQ, L, 2]
    const float* value = (const float*)d_in[2];   // [LV, B, C]
    // d_in[3] spatial_shapes, d_in[4] level_start_index: compile-time constants
    const float* Wv    = (const float*)d_in[5];
    const float* bv    = (const float*)d_in[6];
    const float* Woff  = (const float*)d_in[7];
    const float* boff  = (const float*)d_in[8];
    const float* Wattn = (const float*)d_in[9];
    const float* battn = (const float*)d_in[10];
    const float* Wo    = (const float*)d_in[11];
    const float* bo    = (const float*)d_in[12];
    float* out = (float*)d_out;

    // workspace layout (floats, all sub-offsets 32B aligned)
    float* ws = (float*)d_ws;
    const size_t VAL_F = (size_t)B_N * NH_N * LV_N * D_N;    // slot kept f32-sized
    const size_t LOC_F = (size_t)B_N * LQ_N * NH_N * 32;
    const size_t LGT_F = (size_t)B_N * LQ_N * NH_N * 16;
    const size_t TMP_F = (size_t)B_N * LQ_N * C_N;
    __bf16* val_ws  = (__bf16*)ws;            // uses half of its f32 slot
    float* loc_ws   = ws + VAL_F;
    float* logit_ws = loc_ws + LOC_F;
    float* tmp_ws   = logit_ws + LGT_F;

    // packed hi/lo bf16 weights after the float workspace
    const int PK256 = (C_N / 32) * (256 / 16) * 32 * 8;      // 32768 dwords
    const int PK128 = (C_N / 32) * (128 / 16) * 32 * 8;      // 16384 dwords
    unsigned* pk       = (unsigned*)(tmp_ws + TMP_F);
    unsigned* wv_hi    = pk;                 unsigned* wv_lo    = wv_hi + PK256;
    unsigned* woff_hi  = wv_lo + PK256;      unsigned* woff_lo  = woff_hi + PK256;
    unsigned* wattn_hi = woff_lo + PK256;    unsigned* wattn_lo = wattn_hi + PK128;
    unsigned* wo_hi    = wattn_lo + PK128;   unsigned* wo_lo    = wo_hi + PK256;

    hipLaunchKernelGGL(k_pack_w, dim3((PK256 + 255) / 256), dim3(256), 0, stream,
                       Wv, 256, wv_hi, wv_lo);
    hipLaunchKernelGGL(k_pack_w, dim3((PK256 + 255) / 256), dim3(256), 0, stream,
                       Woff, 256, woff_hi, woff_lo);
    hipLaunchKernelGGL(k_pack_w, dim3((PK128 + 255) / 256), dim3(256), 0, stream,
                       Wattn, 128, wattn_hi, wattn_lo);
    hipLaunchKernelGGL(k_pack_w, dim3((PK256 + 255) / 256), dim3(256), 0, stream,
                       Wo, 256, wo_hi, wo_lo);

    const int M  = B_N * LQ_N;                               // == B*LV = 26588
    const int MT = (M + 15) / 16;                            // 1662 m-tiles

    hipLaunchKernelGGL(k_val_proj, dim3(MT), dim3(256), 0, stream,
                       value, wv_hi, wv_lo, bv, val_ws);

    hipLaunchKernelGGL(k_qproj, dim3(MT), dim3(256), 0, stream,
                       query, woff_hi, woff_lo, wattn_hi, wattn_lo,
                       boff, battn, rp, loc_ws, logit_ws);

    int waves = B_N * LQ_N * NH_N;
    hipLaunchKernelGGL(k_sample, dim3((waves + 7) / 8), dim3(256), 0, stream,
                       val_ws, loc_ws, logit_ws, tmp_ws);

    hipLaunchKernelGGL(k_out_proj, dim3(MT), dim3(256), 0, stream,
                       tmp_ws, wo_hi, wo_lo, bo, query, out);
}
